// Mammoth2DecoderLayer_13434657702335
// MI455X (gfx1250) — matrix operand order repaired and checked by reading, not compiled, not run on hardware
//
#include <hip/hip_runtime.h>
#include <hip/hip_bf16.h>

// ---------------- problem constants ----------------
#define TT   3072
#define DD   2048
#define HH   16
#define KVH  4
#define HDD  128
#define II   5632
#define QKVN ((HH + 2 * KVH) * HDD)   // 3072

typedef __attribute__((ext_vector_type(16))) __bf16 v16bf;
typedef __attribute__((ext_vector_type(8)))  float  v8f;

static __device__ __forceinline__ __bf16 f2bf(float f) {
  unsigned int u = __builtin_bit_cast(unsigned int, f);
  unsigned int r = u + 0x7FFFu + ((u >> 16) & 1u);   // round-to-nearest-even
  unsigned short s = (unsigned short)(r >> 16);
  return __builtin_bit_cast(__bf16, s);
}

union FragU {
  uint4 u[2];
  v16bf v;
};
union Pack16 {
  __bf16 h[16];
  uint4  u[2];
};

// two 16B chunks -> one 16x16x32 bf16 fragment (forces ds/global b128 loads)
static __device__ __forceinline__ v16bf ld_frag2(const __bf16* p0, const __bf16* p1) {
  FragU f;
  f.u[0] = *(const uint4*)p0;
  f.u[1] = *(const uint4*)p1;
  return f.v;
}
static __device__ __forceinline__ v16bf ld_frag16(const __bf16* p) {
  return ld_frag2(p, p + 8);
}

// ---- CDNA5 async global->LDS staging (ASYNCcnt) ----
static __device__ __forceinline__ unsigned lds_addr(const void* p) {
  // LDS aperture: addr[31:0] is the LDS byte offset
  return (unsigned)(unsigned long long)p;
}
static __device__ __forceinline__ void async_load_b128(unsigned lds_off, const void* g) {
  asm volatile("global_load_async_to_lds_b128 %0, %1, off"
               :: "v"(lds_off), "v"(g) : "memory");
}
static __device__ __forceinline__ void wait_async0() {
  asm volatile("s_wait_asynccnt 0x0" ::: "memory");
}

// ======================================================================
// Fused (a [+ b]) -> residual fp32 out (optional) + RMSNorm * w -> bf16
// ======================================================================
__global__ __launch_bounds__(256) void add_rmsnorm_kernel(
    const float* __restrict__ a, const float* __restrict__ b,
    const float* __restrict__ w, float* __restrict__ sum_out,
    __bf16* __restrict__ norm_out) {
  __shared__ float red[8];
  __shared__ float inv_s;
  const int t = blockIdx.x;
  const float* ar = a + (size_t)t * DD;
  const float* br = b ? b + (size_t)t * DD : nullptr;
  float vals[8];
  float ss = 0.f;
#pragma unroll
  for (int i = 0; i < 8; i++) {
    int c = threadIdx.x + i * 256;
    float v = ar[c] + (br ? br[c] : 0.f);
    vals[i] = v;
    ss += v * v;
  }
#pragma unroll
  for (int o = 16; o > 0; o >>= 1) ss += __shfl_down(ss, o, 32);
  if ((threadIdx.x & 31) == 0) red[threadIdx.x >> 5] = ss;
  __syncthreads();
  if (threadIdx.x == 0) {
    float s = 0.f;
#pragma unroll
    for (int i = 0; i < 8; i++) s += red[i];
    inv_s = rsqrtf(s / (float)DD + 1e-6f);
  }
  __syncthreads();
  const float inv = inv_s;
#pragma unroll
  for (int i = 0; i < 8; i++) {
    int c = threadIdx.x + i * 256;
    if (sum_out) sum_out[(size_t)t * DD + c] = vals[i];
    norm_out[(size_t)t * DD + c] = f2bf(vals[i] * inv * w[c]);
  }
}

// ======================================================================
// GEMM: C[M,N] = A_bf16[M,K] @ B_fp32[K,N] (+bias), templated epilogue.
// block tile 128x128, K-step 32, 8 waves, wave tile 32x64 (2x4 WMMA).
// Software-pipelined double-buffered LDS: async A copy + coalesced B
// loads for tile t+1 are issued before the WMMAs of tile t; B is
// converted/stored and ASYNCcnt waited after the WMMAs. One barrier
// per K-step.
//   EPI 0: outf = acc (+bias)
//   EPI 1: outh = bf16( silu(aux) * acc )
//   EPI 2: outf = acc + aux
//   EPI 3: outf = acc where rowmask[row]!=0
// ======================================================================
template <int EPI>
__global__ __launch_bounds__(256) void gemm_bf16_kernel(
    const __bf16* __restrict__ A, const float* __restrict__ B,
    const float* __restrict__ bias, const float* __restrict__ aux,
    const unsigned char* __restrict__ rowmask,
    float* __restrict__ outf, __bf16* __restrict__ outh,
    int M, int N, int K) {
  __shared__ __bf16 Asub[2][128][40];  // [m][k], rows 80B (16B aligned)
  __shared__ __bf16 BT[2][128][40];    // [n][k], K-contiguous

  const int tid  = threadIdx.x;
  const int lane = tid & 31;
  const int wave = tid >> 5;
  const int bm = blockIdx.y * 128;
  const int bn = blockIdx.x * 128;
  const int wm = (wave >> 1) * 32;
  const int wn = (wave & 1) * 64;
  const int ln15 = lane & 15;
  const int kh   = lane >> 4;  // 0/1

  // staging maps
  const int arow = tid >> 1, acb = (tid & 1) * 16;   // A: 16 bf16/thr
  const int bnl = tid & 127, bk0 = (tid >> 7) * 16;  // B: 16 k/thr

  v8f acc[2][4];
#pragma unroll
  for (int i = 0; i < 2; i++)
#pragma unroll
    for (int j = 0; j < 4; j++)
#pragma unroll
      for (int r = 0; r < 8; r++) acc[i][j][r] = 0.f;

  const __bf16* gA0 = A + (size_t)(bm + arow) * K + acb;
  const float* gB0 = B + (size_t)bk0 * N + bn + bnl;

  // ---- prologue: fully stage tile 0 into buffer 0 ----
  {
    unsigned l = lds_addr(&Asub[0][arow][acb]);
    async_load_b128(l, gA0);
    async_load_b128(l + 16, gA0 + 8);
    Pack16 p;
#pragma unroll
    for (int i = 0; i < 16; i++) p.h[i] = f2bf(gB0[(size_t)i * N]);
    uint4* dst = (uint4*)(&BT[0][bnl][bk0]);
    dst[0] = p.u[0];
    dst[1] = p.u[1];
    wait_async0();
  }
  __syncthreads();

  int buf = 0;
  for (int kt = 0; kt < K; kt += 32) {
    const bool has_next = (kt + 32) < K;
    float breg[16];
    if (has_next) {
      // issue next A tile async copy + next B global loads (overlap WMMAs)
      const __bf16* gA = gA0 + kt + 32;
      unsigned l = lds_addr(&Asub[buf ^ 1][arow][acb]);
      async_load_b128(l, gA);
      async_load_b128(l + 16, gA + 8);
      const float* gB = gB0 + (size_t)(kt + 32) * N;
#pragma unroll
      for (int i = 0; i < 16; i++) breg[i] = gB[(size_t)i * N];
    }

    // compute on current buffer
    v16bf afrag[2], bfrag[4];
#pragma unroll
    for (int i = 0; i < 2; i++) {
      const __bf16* ar = &Asub[buf][wm + i * 16 + ln15][0];
      afrag[i] = ld_frag2(ar + kh * 8, ar + 16 + kh * 8);
    }
#pragma unroll
    for (int j = 0; j < 4; j++)
      bfrag[j] = ld_frag2(&BT[buf][wn + j * 16 + ln15][kh * 8], &BT[buf][wn + j * 16 + ln15][16 + kh * 8]);
#pragma unroll
    for (int i = 0; i < 2; i++)
#pragma unroll
      for (int j = 0; j < 4; j++)
        acc[i][j] = __builtin_amdgcn_wmma_f32_16x16x32_bf16(
            false, afrag[i], false, bfrag[j], (short)0, acc[i][j], false, false);

    if (has_next) {
      Pack16 p;
#pragma unroll
      for (int i = 0; i < 16; i++) p.h[i] = f2bf(breg[i]);
      uint4* dst = (uint4*)(&BT[buf ^ 1][bnl][bk0]);
      dst[0] = p.u[0];
      dst[1] = p.u[1];
      wait_async0();
    }
    __syncthreads();
    buf ^= 1;
  }

  const int rh = kh * 8;  // C/D layout: row = r + 8*(lane>=16)
#pragma unroll
  for (int i = 0; i < 2; i++)
#pragma unroll
    for (int j = 0; j < 4; j++) {
      int col = bn + wn + j * 16 + ln15;
      float bv = (EPI == 0 && bias) ? bias[col] : 0.f;
#pragma unroll
      for (int r = 0; r < 8; r++) {
        int row = bm + wm + i * 16 + rh + r;
        float vacc = acc[i][j][r] + bv;
        size_t off = (size_t)row * N + col;
        if (EPI == 0) {
          outf[off] = vacc;
        } else if (EPI == 1) {
          float g = aux[off];
          float s = g / (1.f + __expf(-g));
          outh[off] = f2bf(s * vacc);
        } else if (EPI == 2) {
          outf[off] = vacc + aux[off];
        } else {
          if (rowmask[row]) outf[off] = vacc;
        }
      }
    }
}

// ======================================================================
// QKV split + NeoX RoPE -> bf16 Q/K/V (bias fused into GEMM)
// ======================================================================
__global__ __launch_bounds__(256) void qkv_rope_kernel(
    const float* __restrict__ qkv, const int* __restrict__ pos,
    __bf16* __restrict__ q, __bf16* __restrict__ k, __bf16* __restrict__ v) {
  const int t = blockIdx.x;
  const float p = (float)pos[t];
  const float* row = qkv + (size_t)t * QKVN;
  for (int idx = threadIdx.x; idx < HH * HDD; idx += 256) {
    int h = idx >> 7, d = idx & 127, dd = d & 63;
    float ang = p * __powf(1.0e6f, -(float)(2 * dd) / (float)HDD);
    float s, c;
    __sincosf(ang, &s, &c);
    float x1 = row[h * HDD + dd];
    float x2 = row[h * HDD + 64 + dd];
    float r = (d < 64) ? (x1 * c - x2 * s) : (x2 * c + x1 * s);
    q[(size_t)t * HH * HDD + idx] = f2bf(r);
  }
  const float* krow = row + HH * HDD;
  for (int idx = threadIdx.x; idx < KVH * HDD; idx += 256) {
    int h = idx >> 7, d = idx & 127, dd = d & 63;
    float ang = p * __powf(1.0e6f, -(float)(2 * dd) / (float)HDD);
    float s, c;
    __sincosf(ang, &s, &c);
    float x1 = krow[h * HDD + dd];
    float x2 = krow[h * HDD + 64 + dd];
    float r = (d < 64) ? (x1 * c - x2 * s) : (x2 * c + x1 * s);
    k[(size_t)t * KVH * HDD + idx] = f2bf(r);
  }
  const float* vrow = krow + KVH * HDD;
  for (int idx = threadIdx.x; idx < KVH * HDD; idx += 256)
    v[(size_t)t * KVH * HDD + idx] = f2bf(vrow[idx]);
}

// ======================================================================
// Flash attention (causal, GQA). grid (H, T/128); 8 waves x 16 query rows.
// K tile stored straight [j][d] (async copy; B of Q*K^T is per-lane
// contiguous along d); V tile stored transposed [d][j] (per-lane
// contiguous along keys for P@V). Double-buffered K/V staging pipelined
// against the WMMAs. fp32 online softmax, bf16 WMMA.
// ======================================================================
#define KEYB 32
__global__ __launch_bounds__(256) void attn_kernel(
    const __bf16* __restrict__ q, const __bf16* __restrict__ k,
    const __bf16* __restrict__ v, __bf16* __restrict__ out) {
  __shared__ __bf16 Ks[2][KEYB][HDD + 8];   // [j][d]
  __shared__ __bf16 Vt[2][HDD][KEYB + 8];   // [d][j]
  __shared__ float Sb[8][16 * KEYB];        // per-wave S scratch
  __shared__ __bf16 Pb[8][16][KEYB + 8];    // per-wave P (bf16)
  __shared__ float mst[8][16], lst[8][16], sst[8][16];

  const int h = blockIdx.x;
  const int qb = blockIdx.y;
  const int kvh = h >> 2;  // H/KV = 4
  const int tid = threadIdx.x, lane = tid & 31, wave = tid >> 5;
  const int ln15 = lane & 15, kh = lane >> 4;
  const int q0 = qb * 128 + wave * 16;

  // staging maps: 32 keys x 128 d, 16 bf16 per thread
  const int sj = tid >> 3, sdb = (tid & 7) * 16;
  const __bf16* gK0 = k + ((size_t)sj * KVH + kvh) * HDD + sdb;
  const __bf16* gV0 = v + ((size_t)sj * KVH + kvh) * HDD + sdb;
  const size_t kbStride = (size_t)KEYB * KVH * HDD;  // elems per key block

  // Q fragments: 4 K-chunks of 32 over HD=128 (global b128 loads)
  v16bf qf[4];
  {
    const __bf16* qp = q + ((size_t)(q0 + ln15) * HH + h) * HDD;
#pragma unroll
    for (int c = 0; c < 4; c++)
      qf[c] = ld_frag2(qp + c * 32 + kh * 8, qp + c * 32 + 16 + kh * 8);
  }

  v8f oacc[8];
#pragma unroll
  for (int i = 0; i < 8; i++)
#pragma unroll
    for (int r = 0; r < 8; r++) oacc[i][r] = 0.f;
  if (lane < 16) {
    mst[wave][lane] = -1e30f;
    lst[wave][lane] = 0.f;
  }

  // ---- prologue: stage key block 0 into buffer 0 ----
  {
    unsigned l = lds_addr(&Ks[0][sj][sdb]);
    async_load_b128(l, gK0);
    async_load_b128(l + 16, gK0 + 8);
    FragU f;
    f.u[0] = ((const uint4*)gV0)[0];
    f.u[1] = ((const uint4*)gV0)[1];
#pragma unroll
    for (int i = 0; i < 16; i++) Vt[0][sdb + i][sj] = f.v[i];
    wait_async0();
  }
  __syncthreads();

  const float scale = 0.08838834764831845f;    // 1/sqrt(128)
  const int kb_max = (qb * 128 + 127) / KEYB;  // inclusive
  int buf = 0;
  for (int kb = 0; kb <= kb_max; kb++) {
    const bool has_next = kb < kb_max;
    FragU vreg;
    if (has_next) {  // issue next K async + next V global loads
      const __bf16* gK = gK0 + (size_t)(kb + 1) * kbStride;
      unsigned l = lds_addr(&Ks[buf ^ 1][sj][sdb]);
      async_load_b128(l, gK);
      async_load_b128(l + 16, gK + 8);
      const uint4* gV = (const uint4*)(gV0 + (size_t)(kb + 1) * kbStride);
      vreg.u[0] = gV[0];
      vreg.u[1] = gV[1];
    }

    // S = Q K^T  (2 n-tiles x 4 K-chunks)
    v8f sacc[2];
#pragma unroll
    for (int nt = 0; nt < 2; nt++) {
#pragma unroll
      for (int r = 0; r < 8; r++) sacc[nt][r] = 0.f;
#pragma unroll
      for (int c = 0; c < 4; c++) {
        v16bf bf = ld_frag2(&Ks[buf][nt * 16 + ln15][c * 32 + kh * 8], &Ks[buf][nt * 16 + ln15][c * 32 + 16 + kh * 8]);
        sacc[nt] = __builtin_amdgcn_wmma_f32_16x16x32_bf16(
            false, qf[c], false, bf, (short)0, sacc[nt], false, false);
      }
    }
#pragma unroll
    for (int nt = 0; nt < 2; nt++)
#pragma unroll
      for (int r = 0; r < 8; r++)
        Sb[wave][(kh * 8 + r) * KEYB + nt * 16 + ln15] = sacc[nt][r] * scale;

    // online softmax (2 lanes per row, wave-private LDS)
    {
      int row = lane >> 1;
      int half = lane & 1;
      int qg = q0 + row;
      float mx = -1e30f;
      for (int c2 = 0; c2 < 16; c2++) {
        int col = half * 16 + c2;
        int kg = kb * KEYB + col;
        float sv = Sb[wave][row * KEYB + col];
        if (kg <= qg && sv > mx) mx = sv;
      }
      float mx2 = __shfl_xor(mx, 1, 32);
      mx = mx > mx2 ? mx : mx2;
      float mold = mst[wave][row];
      float mnew = mold > mx ? mold : mx;
      float sc = __expf(mold - mnew);
      float ps = 0.f;
      for (int c2 = 0; c2 < 16; c2++) {
        int col = half * 16 + c2;
        int kg = kb * KEYB + col;
        float pv = (kg <= qg) ? __expf(Sb[wave][row * KEYB + col] - mnew) : 0.f;
        ps += pv;
        Pb[wave][row][col] = f2bf(pv);
      }
      ps += __shfl_xor(ps, 1, 32);
      if (half == 0) {
        mst[wave][row] = mnew;
        lst[wave][row] = lst[wave][row] * sc + ps;
        sst[wave][row] = sc;
      }
    }

    // rescale O, accumulate P @ V
    float rs[8];
#pragma unroll
    for (int r = 0; r < 8; r++) rs[r] = sst[wave][kh * 8 + r];
#pragma unroll
    for (int i = 0; i < 8; i++)
#pragma unroll
      for (int r = 0; r < 8; r++) oacc[i][r] *= rs[r];
    {
      const __bf16* pr = &Pb[wave][ln15][0];
      v16bf pf = ld_frag2(pr + kh * 8, pr + 16 + kh * 8);
#pragma unroll
      for (int i = 0; i < 8; i++) {
        v16bf bf = ld_frag2(&Vt[buf][i * 16 + ln15][kh * 8], &Vt[buf][i * 16 + ln15][16 + kh * 8]);
        oacc[i] = __builtin_amdgcn_wmma_f32_16x16x32_bf16(
            false, pf, false, bf, (short)0, oacc[i], false, false);
      }
    }

    if (has_next) {  // finish staging next tile
#pragma unroll
      for (int i = 0; i < 16; i++) Vt[buf ^ 1][sdb + i][sj] = vreg.v[i];
      wait_async0();
    }
    __syncthreads();
    buf ^= 1;
  }

  float invl[8];
#pragma unroll
  for (int r = 0; r < 8; r++) invl[r] = 1.f / lst[wave][kh * 8 + r];
#pragma unroll
  for (int i = 0; i < 8; i++)
#pragma unroll
    for (int r = 0; r < 8; r++) {
      int row = q0 + kh * 8 + r;
      out[(size_t)row * (HH * HDD) + h * HDD + i * 16 + ln15] =
          f2bf(oacc[i][r] * invl[r]);
    }
}

// ======================================================================
// Host orchestration
// ======================================================================
extern "C" void kernel_launch(void* const* d_in, const int* in_sizes, int n_in,
                              void* d_out, int out_size, void* d_ws,
                              size_t ws_size, hipStream_t stream) {
  const int* positions = (const int*)d_in[0];
  const float* hidden = (const float*)d_in[1];
  const float* residual = (const float*)d_in[2];
  const unsigned char* mask = (const unsigned char*)d_in[3];
  const float* ln1 = (const float*)d_in[4];
  const float* w_qkv = (const float*)d_in[5];
  const float* b_qkv = (const float*)d_in[6];
  const float* w_o = (const float*)d_in[7];
  const float* ln2 = (const float*)d_in[8];
  const float* und_gate = (const float*)d_in[9];
  const float* und_up = (const float*)d_in[10];
  const float* und_down = (const float*)d_in[11];
  const float* gen_gate = (const float*)d_in[12];
  const float* gen_up = (const float*)d_in[13];
  const float* gen_down = (const float*)d_in[14];

  float* out_hs = (float*)d_out;
  float* out_resid = out_hs + (size_t)TT * DD;

  char* ws = (char*)d_ws;
  size_t off = 0;
  auto alloc = [&](size_t bytes) {
    char* p = ws + off;
    off += (bytes + 255) & ~(size_t)255;
    return p;
  };
  float* resid1 = (float*)alloc((size_t)TT * DD * 4);
  __bf16* hsn1 = (__bf16*)alloc((size_t)TT * DD * 2);
  float* qkvbuf = (float*)alloc((size_t)TT * QKVN * 4);
  __bf16* qb = (__bf16*)alloc((size_t)TT * HH * HDD * 2);
  __bf16* kb = (__bf16*)alloc((size_t)TT * KVH * HDD * 2);
  __bf16* vb = (__bf16*)alloc((size_t)TT * KVH * HDD * 2);
  __bf16* attnb = (__bf16*)alloc((size_t)TT * HH * HDD * 2);
  __bf16* hsn2 = (__bf16*)alloc((size_t)TT * DD * 2);
  float* gatebuf = (float*)alloc((size_t)TT * II * 4);
  __bf16* hmlp = (__bf16*)alloc((size_t)TT * II * 2);

  // 1) x = hidden + residual; resid1 = x; hsn1 = rmsnorm(x)*ln1
  add_rmsnorm_kernel<<<TT, 256, 0, stream>>>(hidden, residual, ln1, resid1, hsn1);

  // 2) qkv = hsn1 @ w_qkv + b_qkv
  gemm_bf16_kernel<0><<<dim3(QKVN / 128, TT / 128), 256, 0, stream>>>(
      hsn1, w_qkv, b_qkv, nullptr, nullptr, qkvbuf, nullptr, TT, QKVN, DD);

  // 3) split + RoPE -> bf16 q/k/v
  qkv_rope_kernel<<<TT, 256, 0, stream>>>(qkvbuf, positions, qb, kb, vb);

  // 4) causal GQA attention
  attn_kernel<<<dim3(HH, TT / 128), 256, 0, stream>>>(qb, kb, vb, attnb);

  // 5) resid2 = attn @ w_o + resid1   (written straight into output)
  gemm_bf16_kernel<2><<<dim3(DD / 128, TT / 128), 256, 0, stream>>>(
      attnb, w_o, nullptr, resid1, nullptr, out_resid, nullptr, TT, DD, DD);

  // 6) hsn2 = rmsnorm(resid2)*ln2
  add_rmsnorm_kernel<<<TT, 256, 0, stream>>>(out_resid, nullptr, ln2, nullptr, hsn2);

  // 7) und expert: gate, up (fused silu*up -> bf16), down -> out_hs (all rows)
  gemm_bf16_kernel<0><<<dim3(II / 128, TT / 128), 256, 0, stream>>>(
      hsn2, und_gate, nullptr, nullptr, nullptr, gatebuf, nullptr, TT, II, DD);
  gemm_bf16_kernel<1><<<dim3(II / 128, TT / 128), 256, 0, stream>>>(
      hsn2, und_up, nullptr, gatebuf, nullptr, nullptr, hmlp, TT, II, DD);
  gemm_bf16_kernel<0><<<dim3(DD / 128, TT / 128), 256, 0, stream>>>(
      hmlp, und_down, nullptr, nullptr, nullptr, out_hs, nullptr, TT, DD, II);

  // 8) gen expert: same, but down-proj overwrites only masked rows
  gemm_bf16_kernel<0><<<dim3(II / 128, TT / 128), 256, 0, stream>>>(
      hsn2, gen_gate, nullptr, nullptr, nullptr, gatebuf, nullptr, TT, II, DD);
  gemm_bf16_kernel<1><<<dim3(II / 128, TT / 128), 256, 0, stream>>>(
      hsn2, gen_up, nullptr, gatebuf, nullptr, nullptr, hmlp, TT, II, DD);
  gemm_bf16_kernel<3><<<dim3(DD / 128, TT / 128), 256, 0, stream>>>(
      hmlp, gen_down, nullptr, nullptr, mask, out_hs, nullptr, TT, DD, II);
}
